// Attention_17076789969152
// MI455X (gfx1250) — matrix operand order repaired and checked by reading, not compiled, not run on hardware
//
#include <hip/hip_runtime.h>
#include <hip/hip_bf16.h>
#include <stdint.h>

// ---------------------------------------------------------------------------
// Problem constants (from reference)
// ---------------------------------------------------------------------------
#define BATCH   2
#define SEQ     2048
#define DIM     1024
#define HEADS   16
#define DHEAD   64
#define INNER   1024            // HEADS * DHEAD
#define ROWS    (BATCH * SEQ)   // 4096
#define SCALE   0.125f          // DHEAD ** -0.5

typedef __attribute__((ext_vector_type(16))) __bf16 v16bf;
typedef __attribute__((ext_vector_type(8)))  __bf16 v8bf;
typedef __attribute__((ext_vector_type(8)))  float  v8f;
typedef __attribute__((ext_vector_type(4)))  unsigned int u32x4;
typedef __attribute__((ext_vector_type(8)))  int i32x8;
typedef __attribute__((ext_vector_type(4)))  int i32x4;

union ABFrag { v16bf v; v8bf h[2]; };

__device__ __forceinline__ v8f wmma_bf16(v16bf a, v16bf b, v8f c) {
    return __builtin_amdgcn_wmma_f32_16x16x32_bf16(
        /*neg_a=*/false, a, /*neg_b=*/false, b,
        /*c_mod=*/(short)0, c, /*reuse_a=*/false, /*reuse_b=*/false);
}

// ---------------------------------------------------------------------------
// TDM: async DMA of a 2-D bf16 tile (tile_d0 elems wide, tile_d1 rows, row
// stride stride_d0 elems) from global to LDS. Descriptor layout per
// CDNA5 ISA (D# group0 + group1, groups 2/3 unused for 2D).
// ---------------------------------------------------------------------------
__device__ __forceinline__ void tdm_load_2d_bf16(const void* gaddr,
                                                 unsigned lds_off,
                                                 unsigned tile_d0, unsigned tile_d1,
                                                 unsigned tensor_d0, unsigned tensor_d1,
                                                 unsigned stride_d0)
{
    unsigned long long ga = (unsigned long long)(uintptr_t)gaddr;
    u32x4 g0;
    g0[0] = 1u;                                         // count=1 (valid D#)
    g0[1] = lds_off;                                    // LDS byte address
    g0[2] = (unsigned)(ga & 0xFFFFFFFFu);               // global_addr[31:0]
    g0[3] = (unsigned)((ga >> 32) & 0x01FFFFFFu)        // global_addr[56:32]
          | (2u << 30);                                 // type=2 (image)
    i32x8 g1;
    g1[0] = (int)(1u << 16);                            // data_size=1 -> 2 bytes
    g1[1] = (int)((tensor_d0 & 0xFFFFu) << 16);         // tensor_dim0[15:0]
    g1[2] = (int)((tensor_d0 >> 16) | ((tensor_d1 & 0xFFFFu) << 16));
    g1[3] = (int)((tensor_d1 >> 16) | (tile_d0 << 16)); // tile_dim0
    g1[4] = (int)(tile_d1 & 0xFFFFu);                   // tile_dim1 (tile_dim2=0)
    g1[5] = (int)stride_d0;                             // tensor_dim0_stride[31:0]
    g1[6] = 0;
    g1[7] = 0;
    i32x4 z4 = {};
    i32x8 z8 = {};
    __builtin_amdgcn_tensor_load_to_lds(g0, g1, z4, z4, z8, 0);
}

// ---------------------------------------------------------------------------
// 1) LayerNorm: fp32 in -> bf16 out. One wave32 per row of 1024.
// ---------------------------------------------------------------------------
__global__ __launch_bounds__(256)
void ln_bf16_kernel(const float* __restrict__ x,
                    const float* __restrict__ gamma,
                    const float* __restrict__ beta,
                    __bf16* __restrict__ xn)
{
    const int wave = threadIdx.x >> 5;
    const int lane = threadIdx.x & 31;
    const int row  = blockIdx.x * 8 + wave;
    if (row >= ROWS) return;

    const float* xr = x + (size_t)row * DIM;
    float v[32];
    float s = 0.f, ss = 0.f;
#pragma unroll
    for (int i = 0; i < 32; ++i) {
        float t = xr[i * 32 + lane];
        v[i] = t; s += t; ss += t * t;
    }
#pragma unroll
    for (int off = 16; off > 0; off >>= 1) {
        s  += __shfl_xor(s,  off, 32);
        ss += __shfl_xor(ss, off, 32);
    }
    const float mean = s * (1.0f / DIM);
    const float var  = ss * (1.0f / DIM) - mean * mean;
    const float rstd = rsqrtf(var + 1e-5f);

    __bf16* out = xn + (size_t)row * DIM;
#pragma unroll
    for (int i = 0; i < 32; ++i) {
        int idx = i * 32 + lane;
        out[idx] = (__bf16)((v[i] - mean) * rstd * gamma[idx] + beta[idx]);
    }
}

// ---------------------------------------------------------------------------
// 2) Weight pack: fp32 W[K,N] row-major -> bf16 B-fragment layout.
// ---------------------------------------------------------------------------
__global__ __launch_bounds__(256)
void pack_w_kernel(const float* __restrict__ W, __bf16* __restrict__ Wp,
                   int K, int N)
{
    size_t idx = (size_t)blockIdx.x * blockDim.x + threadIdx.x;
    size_t total = (size_t)K * N;
    if (idx >= total) return;
    int t     = (int)(idx & 15);
    int lane  = (int)((idx >> 4) & 31);
    size_t rest = idx >> 9;
    int ktcnt = K >> 5;
    int kt    = (int)(rest % ktcnt);
    int nt    = (int)(rest / ktcnt);
    int k = kt * 32 + (lane >> 4) * 16 + t;
    int n = nt * 16 + (lane & 15);
    Wp[idx] = (__bf16)W[(size_t)k * N + n];
}

// ---------------------------------------------------------------------------
// 3) GEMM: C[M,N] = A[M,K](bf16 row-major) @ Bpacked(bf16) [+ bias]
//    8 waves/block, block tile 128x128. A tile (128x32) DMA'd into LDS by
//    the TDM (double-buffered, wave 0, s_wait_tensorcnt). Per 32-K step:
//    1 A-frag (2x ds_load_b128) feeds 8 WMMA vs 8 packed-B fragments.
// ---------------------------------------------------------------------------
template <bool F32OUT>
__global__ __launch_bounds__(256)
void gemm_bf16_kernel(const __bf16* __restrict__ A, int lda,
                      const __bf16* __restrict__ Bp,
                      float* __restrict__ Cf, __bf16* __restrict__ Cb, int ldc,
                      const float* __restrict__ bias,
                      int M, int K)
{
    __shared__ __bf16 atile[2][128 * 32];   // 2 x 8 KB

    const int wave  = threadIdx.x >> 5;
    const int lane  = threadIdx.x & 31;
    const int lo    = lane & 15;
    const int khalf = lane >> 4;
    const int m0    = blockIdx.x * 128;
    const int mtile = blockIdx.x * 8 + wave;
    const int nt0   = blockIdx.y * 8;          // 8 n-tiles of 16 = 128 cols
    const int ktcnt = K >> 5;

    const unsigned lds0 = (unsigned)(uintptr_t)(void*)&atile[0][0];
    const unsigned lds1 = (unsigned)(uintptr_t)(void*)&atile[1][0];

    if (wave == 0) {
        tdm_load_2d_bf16(A + (size_t)m0 * lda, lds0, 32, 128,
                         (unsigned)K, (unsigned)M, (unsigned)lda);
    }

    v8f acc[8] = {};

    for (int kt = 0; kt < ktcnt; ++kt) {
        if (wave == 0) {
            if (kt + 1 < ktcnt) {
                tdm_load_2d_bf16(A + (size_t)m0 * lda + (kt + 1) * 32,
                                 ((kt + 1) & 1) ? lds1 : lds0, 32, 128,
                                 (unsigned)K, (unsigned)M, (unsigned)lda);
                __builtin_amdgcn_s_wait_tensorcnt(1);   // tile kt complete
            } else {
                __builtin_amdgcn_s_wait_tensorcnt(0);
            }
        }
        __syncthreads();

        const __bf16* as = &atile[kt & 1][(wave * 16 + lo) * 32];
        ABFrag a;
        a.h[0] = *(const v8bf*)(as + khalf * 16);
        a.h[1] = *(const v8bf*)(as + khalf * 16 + 8);

        if (kt + 1 < ktcnt)   // hint next K-step's weight fragments into cache
            __builtin_prefetch(Bp + ((size_t)(nt0 * ktcnt + kt + 1) * 512), 0, 0);

#pragma unroll
        for (int c = 0; c < 8; ++c) {
            const int nt = nt0 + c;
            const v16bf b = *(const v16bf*)(
                Bp + ((size_t)(nt * ktcnt + kt) * 32 + lane) * 16);
            acc[c] = wmma_bf16(a.v, b, acc[c]);
        }
        __syncthreads();
    }

    // C layout: VGPR r, lane -> (row = r + 8*(lane>>4), col = lane&15)
#pragma unroll
    for (int c = 0; c < 8; ++c) {
        const int n = (nt0 + c) * 16 + lo;
#pragma unroll
        for (int r = 0; r < 8; ++r) {
            const int row = mtile * 16 + r + 8 * khalf;
            float val = acc[c][r];
            if (F32OUT) {
                Cf[(size_t)row * ldc + n] = val + bias[n];
            } else {
                Cb[(size_t)row * ldc + n] = (__bf16)val;
            }
        }
    }
}

// ---------------------------------------------------------------------------
// 4) Pack V (columns INNER..2*INNER of kv) into B-fragment layout for PV WMMA.
// ---------------------------------------------------------------------------
__global__ __launch_bounds__(256)
void pack_v_kernel(const __bf16* __restrict__ KV, __bf16* __restrict__ Vp)
{
    size_t idx = (size_t)blockIdx.x * blockDim.x + threadIdx.x;
    const size_t total = (size_t)BATCH * HEADS * (SEQ / 32) * 4 * 32 * 16;
    if (idx >= total) return;
    int t     = (int)(idx & 15);
    int lane  = (int)((idx >> 4) & 31);
    int chunk = (int)((idx >> 9) & 3);
    size_t rest = idx >> 11;
    int kt32 = (int)(rest % (SEQ / 32));
    int bh   = (int)(rest / (SEQ / 32));
    int h = bh % HEADS;
    int b = bh / HEADS;
    int key = kt32 * 32 + (lane >> 4) * 16 + t;
    int d   = chunk * 16 + (lane & 15);
    Vp[idx] = KV[((size_t)(b * SEQ + key)) * (2 * INNER) + INNER + h * DHEAD + d];
}

// ---------------------------------------------------------------------------
// 5) Flash attention. Block = 8 waves sharing one (b,h); wave w owns query
//    tile blkq*8+w. Per 64-key iteration the K tile (64x64, strided) and the
//    packed V tile (8 KB contiguous) are TDM-DMA'd into LDS, double-buffered,
//    so K/V global traffic is amortized 8x across the block. Per iteration:
//    QK^T 8 WMMA (B-frags from LDS K tile) -> online softmax -> P via LDS
//    re-layout -> PV 8 WMMA (B-frags from LDS V tile).
// ---------------------------------------------------------------------------
__global__ __launch_bounds__(256)
void flash_attn_kernel(const __bf16* __restrict__ Q,    // [ROWS, INNER]
                       const __bf16* __restrict__ KV,   // [ROWS, 2*INNER], K at col 0
                       const __bf16* __restrict__ Vp,   // packed V
                       __bf16* __restrict__ O)          // [ROWS, INNER]
{
    __shared__ __bf16 ktile[2][64 * 64];   // 2 x 8 KB : 64 keys x 64 d
    __shared__ __bf16 vtile[2][4096];      // 2 x 8 KB : packed V fragments
    __shared__ __bf16 p_lds[8][16 * 64];   // 16 KB   : per-wave P tiles

    const int wave = threadIdx.x >> 5;
    const int lane = threadIdx.x & 31;
    const int bh   = blockIdx.x >> 4;                // 0..31 = b*HEADS + h
    const int blkq = blockIdx.x & 15;                // 16 q-groups of 8 tiles
    const int h    = bh & (HEADS - 1);
    const int b    = bh >> 4;
    const int qt   = blkq * 8 + wave;                // this wave's query tile

    const int lo    = lane & 15;
    const int khalf = lane >> 4;

    const unsigned kl0 = (unsigned)(uintptr_t)(void*)&ktile[0][0];
    const unsigned kl1 = (unsigned)(uintptr_t)(void*)&ktile[1][0];
    const unsigned vl0 = (unsigned)(uintptr_t)(void*)&vtile[0][0];
    const unsigned vl1 = (unsigned)(uintptr_t)(void*)&vtile[1][0];

    const __bf16* kbase = KV + (size_t)(b * SEQ) * (2 * INNER) + h * DHEAD;
    const __bf16* vbase = Vp + (size_t)bh * (SEQ / 32) * 2048;

    // Q A-fragments for d-steps 0..31 and 32..63 (loaded once, per wave)
    const __bf16* qbase =
        Q + ((size_t)(b * SEQ + qt * 16 + lo)) * INNER + h * DHEAD;
    ABFrag aq[2];
    aq[0].h[0] = *(const v8bf*)(qbase + 0  + khalf * 16);
    aq[0].h[1] = *(const v8bf*)(qbase + 8  + khalf * 16);
    aq[1].h[0] = *(const v8bf*)(qbase + 32 + khalf * 16);
    aq[1].h[1] = *(const v8bf*)(qbase + 40 + khalf * 16);

    if (wave == 0) {
        tdm_load_2d_bf16(kbase, kl0, 64, 64, 2048, SEQ, 2048);   // K tile 0
        tdm_load_2d_bf16(vbase, vl0, 4096, 1, 4096, 1, 4096);    // V tile 0
    }

    v8f acc[4] = {};
    float rmax[8], rsum[8];
#pragma unroll
    for (int r = 0; r < 8; ++r) { rmax[r] = -1e30f; rsum[r] = 0.f; }

    const int ktcnt = SEQ / 64;
    for (int kt = 0; kt < ktcnt; ++kt) {
        if (wave == 0) {
            if (kt + 1 < ktcnt) {
                const int nb = (kt + 1) & 1;
                tdm_load_2d_bf16(kbase + (size_t)(kt + 1) * 64 * (2 * INNER),
                                 nb ? kl1 : kl0, 64, 64, 2048, SEQ, 2048);
                tdm_load_2d_bf16(vbase + (size_t)(kt + 1) * 4096,
                                 nb ? vl1 : vl0, 4096, 1, 4096, 1, 4096);
                __builtin_amdgcn_s_wait_tensorcnt(2);   // kt's two DMAs done
            } else {
                __builtin_amdgcn_s_wait_tensorcnt(0);
            }
        }
        __syncthreads();

        const __bf16* ks = &ktile[kt & 1][0];
        const __bf16* vs = &vtile[kt & 1][0];

        // ---- scores for 64 keys: four 16-key C tiles ------------------
        v8f dots[4];
#pragma unroll
        for (int s2 = 0; s2 < 4; ++s2) {
            const __bf16* kr = ks + (s2 * 16 + lo) * 64;     // key row in LDS
            v16bf bk0 = *(const v16bf*)(kr + khalf * 16);        // d 0..31
            v16bf bk1 = *(const v16bf*)(kr + 32 + khalf * 16);   // d 32..63
            v8f c = {};
            c = wmma_bf16(aq[0].v, bk0, c);
            c = wmma_bf16(aq[1].v, bk1, c);
            dots[s2] = c;
        }

        // ---- online softmax (rows r+8*khalf, cols across 16 lanes) ----
#pragma unroll
        for (int r = 0; r < 8; ++r) {
            float tmax = fmaxf(fmaxf(dots[0][r], dots[1][r]),
                               fmaxf(dots[2][r], dots[3][r])) * SCALE;
#pragma unroll
            for (int off = 1; off < 16; off <<= 1)
                tmax = fmaxf(tmax, __shfl_xor(tmax, off, 32));
            const float newmax = fmaxf(rmax[r], tmax);
            const float alpha  = __expf(rmax[r] - newmax);
            rmax[r] = newmax;

            const int m = r + 8 * khalf;
            float srow = 0.f;
#pragma unroll
            for (int s2 = 0; s2 < 4; ++s2) {
                const float p = __expf(dots[s2][r] * SCALE - newmax);
                p_lds[wave][m * 64 + s2 * 16 + lo] = (__bf16)p;
                srow += p;
            }
#pragma unroll
            for (int off = 1; off < 16; off <<= 1)
                srow += __shfl_xor(srow, off, 32);
            rsum[r] = rsum[r] * alpha + srow;
#pragma unroll
            for (int c = 0; c < 4; ++c)
                acc[c][r] *= alpha;
        }
        __syncthreads();

        // ---- P (16x64) @ V (64x64): A-frags + B-frags from LDS --------
        ABFrag ap0, ap1;
        ap0.h[0] = *(const v8bf*)(&p_lds[wave][lo * 64 +      khalf * 16]);
        ap0.h[1] = *(const v8bf*)(&p_lds[wave][lo * 64 + 8  + khalf * 16]);
        ap1.h[0] = *(const v8bf*)(&p_lds[wave][lo * 64 + 32 + khalf * 16]);
        ap1.h[1] = *(const v8bf*)(&p_lds[wave][lo * 64 + 40 + khalf * 16]);

#pragma unroll
        for (int c = 0; c < 4; ++c) {
            v16bf bv0 = *(const v16bf*)(vs + (c * 32 + lane) * 16);
            v16bf bv1 = *(const v16bf*)(vs + 2048 + (c * 32 + lane) * 16);
            acc[c] = wmma_bf16(ap0.v, bv0, acc[c]);
            acc[c] = wmma_bf16(ap1.v, bv1, acc[c]);
        }
        __syncthreads();   // all waves done with ktile/vtile[kt&1] before reuse
    }

    // ---- finalize: divide by row sums, store bf16 [b, n, h*64+d] ------
#pragma unroll
    for (int c = 0; c < 4; ++c) {
#pragma unroll
        for (int r = 0; r < 8; ++r) {
            const int row = qt * 16 + r + 8 * khalf;
            const int col = h * DHEAD + c * 16 + lo;
            const float o = acc[c][r] / rsum[r];
            O[((size_t)(b * SEQ + row)) * INNER + col] = (__bf16)o;
        }
    }
}

// ---------------------------------------------------------------------------
// Launch
// ---------------------------------------------------------------------------
extern "C" void kernel_launch(void* const* d_in, const int* in_sizes, int n_in,
                              void* d_out, int out_size, void* d_ws, size_t ws_size,
                              hipStream_t stream)
{
    (void)in_sizes; (void)n_in; (void)out_size; (void)ws_size;

    const float* x     = (const float*)d_in[0];
    const float* w_q   = (const float*)d_in[1];
    const float* w_kv  = (const float*)d_in[2];
    const float* w_out = (const float*)d_in[3];
    const float* b_out = (const float*)d_in[4];
    const float* gamma = (const float*)d_in[5];
    const float* beta  = (const float*)d_in[6];
    float* out = (float*)d_out;

    // workspace carve (bf16 regions, all 256B aligned)
    uint8_t* ws = (uint8_t*)d_ws;
    const size_t MB = 1024ull * 1024ull;
    __bf16* xn_bf   = (__bf16*)(ws + 0);        //  8 MB : [4096,1024]
    __bf16* q_bf    = (__bf16*)(ws + 8 * MB);   //  8 MB : [4096,1024]
    __bf16* kv_bf   = (__bf16*)(ws + 16 * MB);  // 16 MB : [4096,2048]
    __bf16* attn_bf = (__bf16*)(ws + 32 * MB);  //  8 MB : [4096,1024]
    __bf16* wq_p    = (__bf16*)(ws + 40 * MB);  //  2 MB
    __bf16* wkv_p   = (__bf16*)(ws + 42 * MB);  //  4 MB
    __bf16* wout_p  = (__bf16*)(ws + 46 * MB);  //  2 MB
    __bf16* v_p     = (__bf16*)(ws + 48 * MB);  //  8 MB

    // weight packing (independent of activations)
    pack_w_kernel<<<(DIM * INNER) / 256, 256, 0, stream>>>(w_q,   wq_p,   DIM,   INNER);
    pack_w_kernel<<<(DIM * 2 * INNER) / 256, 256, 0, stream>>>(w_kv, wkv_p, DIM, 2 * INNER);
    pack_w_kernel<<<(INNER * DIM) / 256, 256, 0, stream>>>(w_out, wout_p, INNER, DIM);

    // layernorm -> bf16 activations
    ln_bf16_kernel<<<ROWS / 8, 256, 0, stream>>>(x, gamma, beta, xn_bf);

    // projections (block tile 128x128)
    gemm_bf16_kernel<false><<<dim3(ROWS / 128, INNER / 128), 256, 0, stream>>>(
        xn_bf, DIM, wq_p, nullptr, q_bf, INNER, nullptr, ROWS, DIM);
    gemm_bf16_kernel<false><<<dim3(ROWS / 128, (2 * INNER) / 128), 256, 0, stream>>>(
        xn_bf, DIM, wkv_p, nullptr, kv_bf, 2 * INNER, nullptr, ROWS, DIM);

    // pack V fragments for the PV matmul
    {
        const size_t total = (size_t)BATCH * HEADS * (SEQ / 32) * 4 * 32 * 16;
        pack_v_kernel<<<(unsigned)(total / 256), 256, 0, stream>>>(kv_bf, v_p);
    }

    // flash attention: 512 blocks x 8 waves; block shares (b,h), 8 q-tiles
    flash_attn_kernel<<<(BATCH * HEADS * (SEQ / 16)) / 8, 256, 0, stream>>>(
        q_bf, kv_bf, v_p, attn_bf);

    // output projection + bias (fp32 out)
    gemm_bf16_kernel<true><<<dim3(ROWS / 128, DIM / 128), 256, 0, stream>>>(
        attn_bf, INNER, wout_p, out, nullptr, DIM, b_out, ROWS, INNER);
}
